// PointConvSceneFlowPWC2_4_36077725286772
// MI455X (gfx1250) — compile-verified, matrix operand order checked by reading, not run on hardware
//
#include <hip/hip_runtime.h>
#include <hip/hip_bf16.h>

// ---------------------------------------------------------------------------
// PointConvSceneFlowPWC forward for gfx1250 (MI455X), wave32 + WMMA bf16.
// GEMMs: 128x64 block tile, 8 waves x (32x32 wave tile) = 4 WMMA per K-step,
// double-buffered LDS (1 barrier/K-step). Weight tiles staged via
// GLOBAL_LOAD_ASYNC_TO_LDS_B128 (ASYNCcnt); activations staged with
// branch-free clamped loads + fp32->bf16 conversion. Irregular ops
// (fps/knn/gather/interp) are L2-resident brute force with LDS staging.
// ---------------------------------------------------------------------------

#define LEAK 0.1f

typedef __bf16  bf16_t;
typedef __attribute__((ext_vector_type(16))) __bf16 v16bf;
typedef __attribute__((ext_vector_type(8)))  float  v8f;

static inline int cd(long a, long b) { return (int)((a + b - 1) / b); }

__device__ __forceinline__ float activate(float x, int mode) {
  if (mode == 1) return x > 0.f ? x : LEAK * x;          // leaky relu
  if (mode == 2) return x > 0.f ? x : 0.f;               // relu
  if (mode == 3) return fminf(fmaxf(x, -200.f), 200.f);  // clip
  return x;
}

// ---------------------------------------------------------------------------
// Weight pre-pass: Wt[n][k] = bf16(W[k][n]), K padded to Kp (mult of 32) with
// zeros so the WMMA K-pad region is exact (A pad is also zero).
// ---------------------------------------------------------------------------
__global__ void convert_wT_kernel(const float* __restrict__ W, int K, int N,
                                  int Kp, bf16_t* __restrict__ Wt) {
  long t = (long)blockIdx.x * blockDim.x + threadIdx.x;
  if (t >= (long)N * Kp) return;
  int n = (int)(t / Kp), k = (int)(t % Kp);
  Wt[t] = (k < K) ? (bf16_t)W[(long)k * N + n] : (bf16_t)0.f;
}

// ---------------------------------------------------------------------------
// WMMA GEMM: C[M,N] = act(A[M,K] @ W[K,N] + b[N]); W passed as bf16 Wt[N][Kp].
// ---------------------------------------------------------------------------
#define BT_M 128
#define BT_N 64
#define BT_K 32
#define A_PAD 8   // row stride 40 elems = 80B: 16B-aligned frag chunks, no conflicts

__device__ __forceinline__ v16bf load_frag(const bf16_t* __restrict__ row, int half) {
  v16bf f;
#pragma unroll
  for (int p = 0; p < 8; ++p) {
    // ISA 7.12.2 16-bit layout: VGPR p holds K pair at
    // K = (p&3)*2 + (p>>2)*16 + 8*half  -> two contiguous 16B chunks
    int kk = ((p & 3) * 2) + ((p >> 2) * 16) + half * 8;
    f[2 * p]     = row[kk];
    f[2 * p + 1] = row[kk + 1];
  }
  return f;
}

__device__ __forceinline__ void store_tile(v8f acc, float* __restrict__ C,
                                           const float* __restrict__ bias,
                                           int M, int N, int row0, int col, int act) {
  if (col >= N) return;
  float bv = bias ? bias[col] : 0.f;
#pragma unroll
  for (int i = 0; i < 8; ++i) {
    int row = row0 + i;
    if (row < M) C[(long)row * N + col] = activate(acc[i] + bv, act);
  }
}

__global__ void gemm_wmma_kernel(const float* __restrict__ A,
                                 const bf16_t* __restrict__ Wt,  // [N][Kp]
                                 const float* __restrict__ bias,
                                 float* __restrict__ C,
                                 int M, int K, int N, int Kp, int act) {
  __shared__ bf16_t Bs[2][BT_N][BT_K];          // [n][k], rows 64B
  __shared__ bf16_t As[2][BT_M][BT_K + A_PAD];  // rows 80B

  const int tid  = threadIdx.x;
  const int wave = tid >> 5;
  const int lane = tid & 31;
  const int bm = blockIdx.x * BT_M;
  const int bn = blockIdx.y * BT_N;
  const int wm = (wave & 3) * 32;   // 4 waves in M
  const int wn = (wave >> 2) * 32;  // 2 waves in N
  const int half = lane >> 4;
  const int lr   = lane & 15;

  // async W staging: thread -> (row wrow of 64, 16B segment wseg of 4)
  const int wrow = tid >> 2;
  const int wseg = tid & 3;
  int gn = bn + wrow;
  if (gn >= N) gn = 0;  // clamp: stay in-bounds, garbage cols are discarded
  const unsigned ldsB0 = (unsigned)(size_t)(void*)&Bs[0][wrow][wseg * 8];
  const unsigned ldsB1 = (unsigned)(size_t)(void*)&Bs[1][wrow][wseg * 8];
  const bf16_t* wbase = Wt + (long)gn * Kp + wseg * 8;

  const int T = (K + BT_K - 1) / BT_K;
  const bool k4 = ((K & 3) == 0);

  auto issueW = [&](int t, unsigned ldsoff) {
    const bf16_t* gp = wbase + (long)t * BT_K;
    asm volatile("global_load_async_to_lds_b128 %0, %1, off"
                 :: "v"(ldsoff), "v"((unsigned long long)(size_t)gp)
                 : "memory");
  };
  auto stageA = [&](int t, bf16_t (*as)[BT_K + A_PAD]) {
    int k0 = t * BT_K;
    if (k4 && (k0 + BT_K) <= K) {
      // vector path: 4 x float4 per thread, branch-free (row-clamped)
#pragma unroll
      for (int it = 0; it < 4; ++it) {
        int t4 = tid + it * 256;            // 0..1023
        int r = t4 >> 3, cg = (t4 & 7) * 4;
        int gr = bm + r;
        const float* src = A + (long)(gr < M ? gr : 0) * K + (k0 + cg);
        float4 v = *(const float4*)src;
        if (gr >= M) { v.x = 0.f; v.y = 0.f; v.z = 0.f; v.w = 0.f; }
        as[r][cg]     = (bf16_t)v.x;
        as[r][cg + 1] = (bf16_t)v.y;
        as[r][cg + 2] = (bf16_t)v.z;
        as[r][cg + 3] = (bf16_t)v.w;
      }
    } else {
      // scalar path: clamped unconditional loads, zeroed by select
#pragma unroll
      for (int it = 0; it < 16; ++it) {
        int tt = tid + it * 256;            // 0..4095
        int r = tt >> 5, cc = tt & 31;
        int gr = bm + r, gc = k0 + cc;
        float v = A[(long)(gr < M ? gr : 0) * K + (gc < K ? gc : 0)];
        if (gr >= M || gc >= K) v = 0.f;
        as[r][cc] = (bf16_t)v;
      }
    }
  };

  v8f acc00 = {0.f,0.f,0.f,0.f,0.f,0.f,0.f,0.f};
  v8f acc01 = acc00, acc10 = acc00, acc11 = acc00;

  // prologue: tile 0 in buffer 0
  issueW(0, ldsB0);
  stageA(0, As[0]);

  for (int t = 0;; ++t) {
    asm volatile("s_wait_asynccnt 0x0" ::: "memory");
    __syncthreads();  // publish buf[t&1]; prior compute done reading buf[(t+1)&1]
    int cur = t & 1;
    if (t + 1 < T) {
      issueW(t + 1, cur ? ldsB0 : ldsB1);
      stageA(t + 1, As[cur ^ 1]);
      __builtin_prefetch(A + (long)(bm + (tid >> 1)) * K + (long)(t + 2) * BT_K, 0, 1);
    }
    v16bf a0 = load_frag(&As[cur][wm + lr][0],      half);
    v16bf a1 = load_frag(&As[cur][wm + 16 + lr][0], half);
    v16bf b0 = load_frag(&Bs[cur][wn + lr][0],      half);
    v16bf b1 = load_frag(&Bs[cur][wn + 16 + lr][0], half);
    acc00 = __builtin_amdgcn_wmma_f32_16x16x32_bf16(false, a0, false, b0, (short)0, acc00, false, false);
    acc01 = __builtin_amdgcn_wmma_f32_16x16x32_bf16(false, a0, false, b1, (short)0, acc01, false, false);
    acc10 = __builtin_amdgcn_wmma_f32_16x16x32_bf16(false, a1, false, b0, (short)0, acc10, false, false);
    acc11 = __builtin_amdgcn_wmma_f32_16x16x32_bf16(false, a1, false, b1, (short)0, acc11, false, false);
    if (t + 1 >= T) break;
  }

  // D layout: VGPR i -> row = i + 8*half, col = lane%16
  store_tile(acc00, C, bias, M, N, bm + wm + half * 8,      bn + wn + lr,      act);
  store_tile(acc01, C, bias, M, N, bm + wm + half * 8,      bn + wn + 16 + lr, act);
  store_tile(acc10, C, bias, M, N, bm + wm + 16 + half * 8, bn + wn + lr,      act);
  store_tile(acc11, C, bias, M, N, bm + wm + 16 + half * 8, bn + wn + 16 + lr, act);
}

// ---------------------------------------------------------------------------
// kNN: one thread per query, reference points chunked through LDS.
// ---------------------------------------------------------------------------
#define KNN_CHUNK 1024

template <int KK>
__global__ void knn_kernel(const float* __restrict__ q, int Nq,
                           const float* __restrict__ ref, int M,
                           int* __restrict__ out) {
  __shared__ float rx[KNN_CHUNK], ry[KNN_CHUNK], rz[KNN_CHUNK];
  int i = blockIdx.x * blockDim.x + threadIdx.x;
  bool valid = i < Nq;
  float qx = 0.f, qy = 0.f, qz = 0.f;
  if (valid) { qx = q[3 * i]; qy = q[3 * i + 1]; qz = q[3 * i + 2]; }
  float bd[KK];
  int   bi[KK];
  for (int j = 0; j < KK; ++j) { bd[j] = 1e30f; bi[j] = 0; }

  for (int c0 = 0; c0 < M; c0 += KNN_CHUNK) {
    int cn = min(KNN_CHUNK, M - c0);
    __syncthreads();
    for (int t = threadIdx.x; t < cn; t += blockDim.x) {
      rx[t] = ref[3 * (c0 + t)];
      ry[t] = ref[3 * (c0 + t) + 1];
      rz[t] = ref[3 * (c0 + t) + 2];
    }
    __syncthreads();
    if (valid) {
      for (int t = 0; t < cn; ++t) {
        float dx = qx - rx[t], dy = qy - ry[t], dz = qz - rz[t];
        float d = dx * dx + dy * dy + dz * dz;
        if (d < bd[KK - 1]) {
          int pos = KK - 1;
          while (pos > 0 && bd[pos - 1] > d) {
            bd[pos] = bd[pos - 1]; bi[pos] = bi[pos - 1]; --pos;
          }
          bd[pos] = d; bi[pos] = c0 + t;
        }
      }
    }
  }
  if (valid)
    for (int j = 0; j < KK; ++j) out[(long)i * KK + j] = bi[j];
}

// ---------------------------------------------------------------------------
// Iterative farthest point sampling (single workgroup, dists in workspace).
// ---------------------------------------------------------------------------
__global__ void fps_kernel(const float* __restrict__ xyz, int N, int S,
                           int* __restrict__ out, float* __restrict__ dists) {
  __shared__ float smax[256];
  __shared__ int   simax[256];
  __shared__ int   s_last;
  int tid = threadIdx.x;
  for (int i = tid; i < N; i += 256) dists[i] = 1e10f;
  if (tid == 0) s_last = 0;
  __syncthreads();
  for (int s = 0; s < S; ++s) {
    int last = s_last;
    if (tid == 0) out[s] = last;
    float lx = xyz[3 * last], ly = xyz[3 * last + 1], lz = xyz[3 * last + 2];
    float bm = -1.f; int bidx = 0x7fffffff;
    for (int i = tid; i < N; i += 256) {
      float dx = xyz[3 * i] - lx, dy = xyz[3 * i + 1] - ly, dz = xyz[3 * i + 2] - lz;
      float d = dx * dx + dy * dy + dz * dz;
      float nd = fminf(dists[i], d);
      dists[i] = nd;
      if (nd > bm) { bm = nd; bidx = i; }
    }
    smax[tid] = bm; simax[tid] = bidx;
    __syncthreads();
    for (int w = 128; w > 0; w >>= 1) {
      if (tid < w) {
        if (smax[tid + w] > smax[tid] ||
            (smax[tid + w] == smax[tid] && simax[tid + w] < simax[tid])) {
          smax[tid] = smax[tid + w]; simax[tid] = simax[tid + w];
        }
      }
      __syncthreads();
    }
    if (tid == 0) s_last = simax[0];
    __syncthreads();
  }
}

// gather rows: out[n,:] = src[idx[n],:]
__global__ void gather_rows_kernel(const float* __restrict__ src, int C,
                                   const int* __restrict__ idx, int Nn,
                                   float* __restrict__ out) {
  long t = (long)blockIdx.x * blockDim.x + threadIdx.x;
  if (t >= (long)Nn * C) return;
  int n = (int)(t / C), ch = (int)(t % C);
  out[t] = src[(long)idx[n] * C + ch];
}

// grouping for pointconv: gx = xyz[idx]-center, gf = [gx | feats[idx]]
__global__ void group_kernel(const float* __restrict__ xyz,
                             const float* __restrict__ feats, int C,
                             const float* __restrict__ centers,
                             const int* __restrict__ idx, int Nn, int K,
                             float* __restrict__ gx, float* __restrict__ gf) {
  long t = (long)blockIdx.x * blockDim.x + threadIdx.x;
  if (t >= (long)Nn * K) return;
  int nl = (int)(t / K), k = (int)(t % K);
  int j = idx[(long)nl * K + k];
  float dx = xyz[3 * j]     - centers[3 * nl];
  float dy = xyz[3 * j + 1] - centers[3 * nl + 1];
  float dz = xyz[3 * j + 2] - centers[3 * nl + 2];
  gx[3 * t] = dx; gx[3 * t + 1] = dy; gx[3 * t + 2] = dz;
  float* o = gf + t * (3 + C);
  o[0] = dx; o[1] = dy; o[2] = dz;
  const float* fp = feats + (long)j * C;
  for (int c = 0; c < C; ++c) o[3 + c] = fp[c];
}

// WeightNet MLP 3->8->8->Cout, relu each layer, one thread per (n,k)
__global__ void weightnet_kernel(const float* __restrict__ d, long NK,
                                 const float* __restrict__ W0, const float* __restrict__ b0,
                                 const float* __restrict__ W1, const float* __restrict__ b1,
                                 const float* __restrict__ W2, const float* __restrict__ b2,
                                 int Cout, float* __restrict__ out) {
  long t = (long)blockIdx.x * blockDim.x + threadIdx.x;
  if (t >= NK) return;
  float x0 = d[3 * t], x1 = d[3 * t + 1], x2 = d[3 * t + 2];
  float h[8], g[8];
#pragma unroll
  for (int i = 0; i < 8; ++i) {
    float v = x0 * W0[i] + x1 * W0[8 + i] + x2 * W0[16 + i] + b0[i];
    h[i] = v > 0.f ? v : 0.f;
  }
#pragma unroll
  for (int i = 0; i < 8; ++i) {
    float v = b1[i];
#pragma unroll
    for (int j = 0; j < 8; ++j) v += h[j] * W1[j * 8 + i];
    g[i] = v > 0.f ? v : 0.f;
  }
  float* o = out + t * Cout;
  for (int c = 0; c < Cout; ++c) {
    float v = b2[c];
#pragma unroll
    for (int j = 0; j < 8; ++j) v += g[j] * W2[j * Cout + c];
    o[c] = v > 0.f ? v : 0.f;
  }
}

// einsum bnkc,bnkw->bn(c*16+w)
__global__ void einsum_kernel(const float* __restrict__ gf,
                              const float* __restrict__ w, int Nn, int K,
                              int C, float* __restrict__ agg) {
  long t = (long)blockIdx.x * blockDim.x + threadIdx.x;
  if (t >= (long)Nn * C) return;
  int n = (int)(t / C), c = (int)(t % C);
  float acc[16];
#pragma unroll
  for (int j = 0; j < 16; ++j) acc[j] = 0.f;
  for (int k = 0; k < K; ++k) {
    float g = gf[((long)n * K + k) * C + c];
    const float* wp = w + ((long)n * K + k) * 16;
#pragma unroll
    for (int j = 0; j < 16; ++j) acc[j] += g * wp[j];
  }
  float* o = agg + ((long)n * C + c) * 16;
#pragma unroll
  for (int j = 0; j < 16; ++j) o[j] = acc[j];
}

// cost-volume grouping: x = [f1[n] | f2[idx] | dir], dirx separately
__global__ void cv_group_kernel(const float* __restrict__ x1,
                                const float* __restrict__ x2,
                                const float* __restrict__ f1,
                                const float* __restrict__ f2, int C1, int C2,
                                const int* __restrict__ idx, int n0, int Nn,
                                int K, float* __restrict__ x,
                                float* __restrict__ dirx) {
  long t = (long)blockIdx.x * blockDim.x + threadIdx.x;
  if (t >= (long)Nn * K) return;
  int nl = (int)(t / K), k = (int)(t % K);
  int n = n0 + nl;
  int j = idx[(long)n * K + k];
  float dx = x2[3 * j]     - x1[3 * n];
  float dy = x2[3 * j + 1] - x1[3 * n + 1];
  float dz = x2[3 * j + 2] - x1[3 * n + 2];
  float* row = x + t * (C1 + C2 + 3);
  const float* a = f1 + (long)n * C1;
  for (int c = 0; c < C1; ++c) row[c] = a[c];
  const float* b = f2 + (long)j * C2;
  for (int c = 0; c < C2; ++c) row[C1 + c] = b[c];
  row[C1 + C2] = dx; row[C1 + C2 + 1] = dy; row[C1 + C2 + 2] = dz;
  dirx[3 * t] = dx; dirx[3 * t + 1] = dy; dirx[3 * t + 2] = dz;
}

// directions only: dir[n,k] = pts[idx[n,k]] - pts[n]   (global n = n0+local)
__global__ void dirs_kernel(const float* __restrict__ pts,
                            const int* __restrict__ idx, int n0, int Nn, int K,
                            float* __restrict__ out) {
  long t = (long)blockIdx.x * blockDim.x + threadIdx.x;
  if (t >= (long)Nn * K) return;
  int nl = (int)(t / K), k = (int)(t % K);
  int n = n0 + nl;
  int j = idx[(long)n * K + k];
  out[3 * t]     = pts[3 * j]     - pts[3 * n];
  out[3 * t + 1] = pts[3 * j + 1] - pts[3 * n + 1];
  out[3 * t + 2] = pts[3 * j + 2] - pts[3 * n + 2];
}

// out[n,c] = sum_k a[n,k,c]*b[n,k,c]
__global__ void mulsum_kernel(const float* __restrict__ a,
                              const float* __restrict__ b, int Nn, int K,
                              int C, float* __restrict__ out) {
  long t = (long)blockIdx.x * blockDim.x + threadIdx.x;
  if (t >= (long)Nn * C) return;
  int n = (int)(t / C), c = (int)(t % C);
  float s = 0.f;
  for (int k = 0; k < K; ++k)
    s += a[((long)n * K + k) * C + c] * b[((long)n * K + k) * C + c];
  out[t] = s;
}

// out[n,c] = sum_k w2[nl,k,c] * p2p[idx[n,k],c]   (n = n0 + nl)
__global__ void p2pgather_kernel(const float* __restrict__ w2,
                                 const int* __restrict__ idx, int n0, int Nn,
                                 int K, int C, const float* __restrict__ p2p,
                                 float* __restrict__ out) {
  long t = (long)blockIdx.x * blockDim.x + threadIdx.x;
  if (t >= (long)Nn * C) return;
  int nl = (int)(t / C), c = (int)(t % C);
  int n = n0 + nl;
  float s = 0.f;
  for (int k = 0; k < K; ++k) {
    int j = idx[(long)n * K + k];
    s += w2[((long)nl * K + k) * C + c] * p2p[(long)j * C + c];
  }
  out[(long)n * C + c] = s;
}

// fused 3-NN inverse-distance interpolation; if base!=null -> out = base - val
__global__ void interp_kernel(const float* __restrict__ dxyz, int Nd,
                              const float* __restrict__ sxyz, int Ms,
                              const float* __restrict__ sf, int C,
                              const float* __restrict__ base,
                              float* __restrict__ out) {
  int i = blockIdx.x * blockDim.x + threadIdx.x;
  if (i >= Nd) return;
  float qx = dxyz[3 * i], qy = dxyz[3 * i + 1], qz = dxyz[3 * i + 2];
  float d0 = 1e30f, d1 = 1e30f, d2 = 1e30f;
  int i0 = 0, i1 = 0, i2 = 0;
  for (int j = 0; j < Ms; ++j) {
    float dx = sxyz[3 * j] - qx, dy = sxyz[3 * j + 1] - qy, dz = sxyz[3 * j + 2] - qz;
    float d = dx * dx + dy * dy + dz * dz;
    if (d < d0)      { d2 = d1; i2 = i1; d1 = d0; i1 = i0; d0 = d; i0 = j; }
    else if (d < d1) { d2 = d1; i2 = i1; d1 = d;  i1 = j; }
    else if (d < d2) { d2 = d;  i2 = j; }
  }
  float w0 = 1.f / fmaxf(sqrtf(d0), 1e-10f);
  float w1 = 1.f / fmaxf(sqrtf(d1), 1e-10f);
  float w2 = 1.f / fmaxf(sqrtf(d2), 1e-10f);
  float ws = w0 + w1 + w2;
  w0 /= ws; w1 /= ws; w2 /= ws;
  for (int c = 0; c < C; ++c) {
    float v = w0 * sf[(long)i0 * C + c] + w1 * sf[(long)i1 * C + c] +
              w2 * sf[(long)i2 * C + c];
    out[(long)i * C + c] = base ? base[(long)i * C + c] - v : v;
  }
}

// channel concat of up to 3 tensors
__global__ void concat3_kernel(const float* __restrict__ a, int Ca,
                               const float* __restrict__ b, int Cb,
                               const float* __restrict__ c, int Cc, int Nn,
                               float* __restrict__ out) {
  int C = Ca + Cb + Cc;
  long t = (long)blockIdx.x * blockDim.x + threadIdx.x;
  if (t >= (long)Nn * C) return;
  int n = (int)(t / C), ch = (int)(t % C);
  float v;
  if (ch < Ca)            v = a[(long)n * Ca + ch];
  else if (ch < Ca + Cb)  v = b[(long)n * Cb + (ch - Ca)];
  else                    v = c[(long)n * Cc + (ch - Ca - Cb)];
  out[t] = v;
}

__global__ void add_kernel(const float* __restrict__ a,
                           const float* __restrict__ b, float* __restrict__ o,
                           long n) {
  long i = (long)blockIdx.x * blockDim.x + threadIdx.x;
  if (i < n) o[i] = a[i] + b[i];
}

// ---------------------------------------------------------------------------
// Host-side orchestration
// ---------------------------------------------------------------------------
struct Lin { const float *W, *b; int cin, cout; };
struct WN3 { Lin a, b, c; };
struct PCD { WN3 wn; Lin lin; };
struct CVP { Lin m0, m1; WN3 wn1, wn2; };
struct FEST { PCD p0, p1; Lin m0, m1, fc, fc2; };

// cursor over flattened params; order follows the reference dict's
// insertion order recursively (each lin -> W then b).
struct Cur {
  void* const* in; int i;
  Lin  L(int ci, int co) { Lin l{(const float*)in[i], (const float*)in[i + 1], ci, co}; i += 2; return l; }
  WN3  WNf(int co)  { WN3 w; w.a = L(3, 8); w.b = L(8, 8); w.c = L(8, co); return w; }
  PCD  PCDf(int ci, int co) { PCD p; p.wn = WNf(16); p.lin = L((ci + 3) * 16, co); return p; }
  CVP  CVf(int c0, int c1, int c2) { CVP v; v.m0 = L(c0, c1); v.m1 = L(c1, c2); v.wn1 = WNf(c2); v.wn2 = WNf(c2); return v; }
  FEST FEf(int ci) { FEST f; f.p0 = PCDf(ci, 128); f.p1 = PCDf(128, 128); f.m0 = L(128, 128); f.m1 = L(128, 64); f.fc = L(64, 3); f.fc2 = L(64, 3); return f; }
};

struct Ctx {
  hipStream_t s;
  char* ws; size_t off, cap;
  char* allocb(size_t bytes) {
    bytes = (bytes + 255) & ~(size_t)255;
    char* p = ws + off; off += bytes; return p;
  }
  float* alloc(size_t elems) { return (float*)allocb(elems * sizeof(float)); }
  int*   alloci(size_t elems) { return (int*)allocb(elems * sizeof(int)); }
};

static const int PC_CHUNK = 2048;
static const int CV_CHUNK = 1024;

static void lin_into(Ctx& c, const float* x, int M, const Lin& L, int act, float* y) {
  int Kp = (L.cin + 31) & ~31;
  size_t mark = c.off;
  bf16_t* Wt = (bf16_t*)c.allocb((size_t)L.cout * Kp * sizeof(bf16_t));
  convert_wT_kernel<<<cd((long)L.cout * Kp, 256), 256, 0, c.s>>>(L.W, L.cin, L.cout, Kp, Wt);
  dim3 g(cd(M, BT_M), cd(L.cout, BT_N));
  gemm_wmma_kernel<<<g, 256, 0, c.s>>>(x, Wt, L.b, y, M, L.cin, L.cout, Kp, act);
  c.off = mark;  // safe: stream is in-order, reuse happens after GEMM runs
}

static float* lin_f(Ctx& c, const float* x, int M, const Lin& L, int act) {
  float* y = c.alloc((size_t)M * L.cout);
  lin_into(c, x, M, L, act, y);
  return y;
}

static float* weightnet_f(Ctx& c, const float* d, long NK, const WN3& w, int Cout) {
  float* o = c.alloc((size_t)NK * Cout);
  weightnet_kernel<<<cd(NK, 256), 256, 0, c.s>>>(d, NK, w.a.W, w.a.b, w.b.W, w.b.b,
                                                 w.c.W, w.c.b, Cout, o);
  return o;
}

// same-resolution continuous conv (self kNN, K=9)
static float* pointconv_f(Ctx& c, const PCD& p, const float* xyz, int N,
                          const float* feats, int C) {
  int* idx = c.alloci((size_t)N * 9);
  knn_kernel<9><<<cd(N, 256), 256, 0, c.s>>>(xyz, N, xyz, N, idx);
  float* out = c.alloc((size_t)N * p.lin.cout);
  for (int n0 = 0; n0 < N; n0 += PC_CHUNK) {
    int nn = min(PC_CHUNK, N - n0);
    size_t mark = c.off;
    float* gx = c.alloc((size_t)nn * 9 * 3);
    float* gf = c.alloc((size_t)nn * 9 * (3 + C));
    group_kernel<<<cd((long)nn * 9, 256), 256, 0, c.s>>>(
        xyz, feats, C, xyz + (size_t)n0 * 3, idx + (size_t)n0 * 9, nn, 9, gx, gf);
    float* w = weightnet_f(c, gx, (long)nn * 9, p.wn, 16);
    float* agg = c.alloc((size_t)nn * (3 + C) * 16);
    einsum_kernel<<<cd((long)nn * (3 + C), 256), 256, 0, c.s>>>(gf, w, nn, 9, 3 + C, agg);
    lin_into(c, agg, nn, p.lin, 1, out + (size_t)n0 * p.lin.cout);
    c.off = mark;
  }
  return out;
}

// FPS downsample + kNN(16) grouping + WeightNet conv
struct PCDOut { float* nxyz; float* feat; };
static PCDOut pointconvd_f(Ctx& c, const PCD& p, const float* xyz, int N,
                           const float* feats, int C, int S) {
  int* fidx = c.alloci(S);
  size_t m0 = c.off;
  float* dists = c.alloc(N);
  fps_kernel<<<1, 256, 0, c.s>>>(xyz, N, S, fidx, dists);
  c.off = m0;
  float* nxyz = c.alloc((size_t)S * 3);
  gather_rows_kernel<<<cd((long)S * 3, 256), 256, 0, c.s>>>(xyz, 3, fidx, S, nxyz);
  int* idx = c.alloci((size_t)S * 16);
  knn_kernel<16><<<cd(S, 256), 256, 0, c.s>>>(nxyz, S, xyz, N, idx);
  float* out = c.alloc((size_t)S * p.lin.cout);
  for (int n0 = 0; n0 < S; n0 += PC_CHUNK) {
    int nn = min(PC_CHUNK, S - n0);
    size_t mark = c.off;
    float* gx = c.alloc((size_t)nn * 16 * 3);
    float* gf = c.alloc((size_t)nn * 16 * (3 + C));
    group_kernel<<<cd((long)nn * 16, 256), 256, 0, c.s>>>(
        xyz, feats, C, nxyz + (size_t)n0 * 3, idx + (size_t)n0 * 16, nn, 16, gx, gf);
    float* w = weightnet_f(c, gx, (long)nn * 16, p.wn, 16);
    float* agg = c.alloc((size_t)nn * (3 + C) * 16);
    einsum_kernel<<<cd((long)nn * (3 + C), 256), 256, 0, c.s>>>(gf, w, nn, 16, 3 + C, agg);
    lin_into(c, agg, nn, p.lin, 1, out + (size_t)n0 * p.lin.cout);
    c.off = mark;
  }
  return {nxyz, out};
}

// PointConvFlow cost volume; C = channels per side
static float* costvol_f(Ctx& c, const CVP& p, const float* x1, int N1,
                        const float* x2, int N2, const float* f1,
                        const float* f2, int C) {
  const int K = 32;
  int Cl = p.m1.cout;
  int* idx = c.alloci((size_t)N1 * K);
  knn_kernel<32><<<cd(N1, 256), 256, 0, c.s>>>(x1, N1, x2, N2, idx);
  float* p2p = c.alloc((size_t)N1 * Cl);
  for (int n0 = 0; n0 < N1; n0 += CV_CHUNK) {
    int nn = min(CV_CHUNK, N1 - n0);
    size_t mark = c.off;
    float* xb  = c.alloc((size_t)nn * K * (2 * C + 3));
    float* dir = c.alloc((size_t)nn * K * 3);
    cv_group_kernel<<<cd((long)nn * K, 256), 256, 0, c.s>>>(
        x1, x2, f1, f2, C, C, idx, n0, nn, K, xb, dir);
    float* h1 = lin_f(c, xb, nn * K, p.m0, 1);
    float* h2 = lin_f(c, h1, nn * K, p.m1, 1);
    float* w1 = weightnet_f(c, dir, (long)nn * K, p.wn1, Cl);
    mulsum_kernel<<<cd((long)nn * Cl, 256), 256, 0, c.s>>>(
        w1, h2, nn, K, Cl, p2p + (size_t)n0 * Cl);
    c.off = mark;
  }
  int* idx2 = c.alloci((size_t)N1 * K);
  knn_kernel<32><<<cd(N1, 256), 256, 0, c.s>>>(x1, N1, x1, N1, idx2);
  float* out = c.alloc((size_t)N1 * Cl);
  for (int n0 = 0; n0 < N1; n0 += CV_CHUNK) {
    int nn = min(CV_CHUNK, N1 - n0);
    size_t mark = c.off;
    float* dir2 = c.alloc((size_t)nn * K * 3);
    dirs_kernel<<<cd((long)nn * K, 256), 256, 0, c.s>>>(x1, idx2, n0, nn, K, dir2);
    float* w2 = weightnet_f(c, dir2, (long)nn * K, p.wn2, Cl);
    p2pgather_kernel<<<cd((long)nn * Cl, 256), 256, 0, c.s>>>(
        w2, idx2, n0, nn, K, Cl, p2p, out);
    c.off = mark;
  }
  return out;
}

static float* interp_f(Ctx& c, const float* dxyz, int Nd, const float* sxyz,
                       int Ms, const float* sf, int C) {
  float* o = c.alloc((size_t)Nd * C);
  interp_kernel<<<cd(Nd, 256), 256, 0, c.s>>>(dxyz, Nd, sxyz, Ms, sf, C, nullptr, o);
  return o;
}

// warp: out = xyz2 - interp(flow1 on xyz1 -> xyz2)
static float* warp_f(Ctx& c, const float* xyz1, int N1, const float* xyz2,
                     int N2, const float* flow1) {
  float* o = c.alloc((size_t)N2 * 3);
  interp_kernel<<<cd(N2, 256), 256, 0, c.s>>>(xyz2, N2, xyz1, N1, flow1, 3, xyz2, o);
  return o;
}

static float* concat2_f(Ctx& c, const float* a, int Ca, const float* b, int Cb, int N) {
  float* o = c.alloc((size_t)N * (Ca + Cb));
  concat3_kernel<<<cd((long)N * (Ca + Cb), 256), 256, 0, c.s>>>(a, Ca, b, Cb, nullptr, 0, N, o);
  return o;
}

struct FEOut { float* feat; float* f; float* ff; };
static FEOut flow_est_f(Ctx& c, const FEST& p, const float* xyz, int N,
                        const float* feats, int Cf, const float* cost, int Cc,
                        const float* flow) {
  int Cin = Cf + Cc + (flow ? 3 : 0);
  float* x = c.alloc((size_t)N * Cin);
  concat3_kernel<<<cd((long)N * Cin, 256), 256, 0, c.s>>>(
      feats, Cf, cost, Cc, flow, flow ? 3 : 0, N, x);
  x = pointconv_f(c, p.p0, xyz, N, x, Cin);
  x = pointconv_f(c, p.p1, xyz, N, x, 128);
  x = lin_f(c, x, N, p.m0, 1);
  x = lin_f(c, x, N, p.m1, 1);   // 64 channels
  float* f  = lin_f(c, x, N, p.fc, 3);   // clip ±200
  float* ff = lin_f(c, x, N, p.fc2, 3);  // clip ±200
  return {x, f, ff};
}

extern "C" void kernel_launch(void* const* d_in, const int* in_sizes, int n_in,
                              void* d_out, int out_size, void* d_ws, size_t ws_size,
                              hipStream_t stream) {
  (void)in_sizes; (void)n_in; (void)out_size;
  const float* xyz1   = (const float*)d_in[0];
  const float* xyz2   = (const float*)d_in[1];
  const float* color1 = (const float*)d_in[2];
  const float* color2 = (const float*)d_in[3];
  const int N = 8192, N1S = 2048, N2S = 512, N3S = 256, N4S = 64;

  Cur cu{d_in, 4};
  Lin l0 = cu.L(3, 32), l0_1 = cu.L(32, 32), l0_2 = cu.L(32, 64);
  PCD lv1 = cu.PCDf(64, 64);   Lin l1_0 = cu.L(64, 64),  l1_1 = cu.L(64, 128);
  PCD lv2 = cu.PCDf(128, 128); Lin l2_0 = cu.L(128, 128), l2_1 = cu.L(128, 256);
  PCD lv3 = cu.PCDf(256, 256); Lin l3_0 = cu.L(256, 256), l3_1 = cu.L(256, 512);
  PCD lv4 = cu.PCDf(512, 256);
  Lin d43 = cu.L(256, 64), d32 = cu.L(256, 64), d21 = cu.L(128, 32), d10 = cu.L(64, 32);
  CVP cv3 = cu.CVf(643, 256, 256), cv2 = cu.CVf(387, 128, 128);
  CVP cv1 = cu.CVf(195, 64, 64),   cv0 = cu.CVf(131, 32, 32);
  FEST fe3 = cu.FEf(512), fe2 = cu.FEf(323), fe1 = cu.FEf(195), fe0 = cu.FEf(131);

  Ctx c{stream, (char*)d_ws, 0, ws_size};

  // ---- level-0 features
  float* t1    = lin_f(c, color1, N, l0, 1);
  float* f1_0  = lin_f(c, t1, N, l0_1, 1);
  float* f1_0b = lin_f(c, f1_0, N, l0_2, 1);
  float* t2    = lin_f(c, color2, N, l0, 1);
  float* f2_0  = lin_f(c, t2, N, l0_1, 1);
  float* f2_0b = lin_f(c, f2_0, N, l0_2, 1);

  // ---- pyramid
  PCDOut d11 = pointconvd_f(c, lv1, xyz1, N, f1_0b, 64, N1S);
  float* f1_1b = lin_f(c, lin_f(c, d11.feat, N1S, l1_0, 1), N1S, l1_1, 1);
  PCDOut d21o = pointconvd_f(c, lv1, xyz2, N, f2_0b, 64, N1S);
  float* f2_1b = lin_f(c, lin_f(c, d21o.feat, N1S, l1_0, 1), N1S, l1_1, 1);

  PCDOut d12 = pointconvd_f(c, lv2, d11.nxyz, N1S, f1_1b, 128, N2S);
  float* f1_2b = lin_f(c, lin_f(c, d12.feat, N2S, l2_0, 1), N2S, l2_1, 1);
  PCDOut d22 = pointconvd_f(c, lv2, d21o.nxyz, N1S, f2_1b, 128, N2S);
  float* f2_2b = lin_f(c, lin_f(c, d22.feat, N2S, l2_0, 1), N2S, l2_1, 1);

  PCDOut d13 = pointconvd_f(c, lv3, d12.nxyz, N2S, f1_2b, 256, N3S);
  float* f1_3b = lin_f(c, lin_f(c, d13.feat, N3S, l3_0, 1), N3S, l3_1, 1);
  PCDOut d23 = pointconvd_f(c, lv3, d22.nxyz, N2S, f2_2b, 256, N3S);
  float* f2_3b = lin_f(c, lin_f(c, d23.feat, N3S, l3_0, 1), N3S, l3_1, 1);

  PCDOut d14 = pointconvd_f(c, lv4, d13.nxyz, N3S, f1_3b, 512, N4S);
  float* f1_4_3 = lin_f(c, interp_f(c, d13.nxyz, N3S, d14.nxyz, N4S, d14.feat, 256), N3S, d43, 1);
  PCDOut d24 = pointconvd_f(c, lv4, d23.nxyz, N3S, f2_3b, 512, N4S);
  float* f2_4_3 = lin_f(c, interp_f(c, d23.nxyz, N3S, d24.nxyz, N4S, d24.feat, 256), N3S, d43, 1);

  // ---- level 3 cost + flow
  float* c1_3 = concat2_f(c, d13.feat, 256, f1_4_3, 64, N3S);   // 320
  float* c2_3 = concat2_f(c, d23.feat, 256, f2_4_3, 64, N3S);
  float* cost3 = costvol_f(c, cv3, d13.nxyz, N3S, d23.nxyz, N3S, c1_3, c2_3, 320);
  FEOut e3 = flow_est_f(c, fe3, d13.nxyz, N3S, d13.feat, 256, cost3, 256, nullptr);

  // ---- cross-level concat features
  float* f1_3_2 = lin_f(c, interp_f(c, d12.nxyz, N2S, d13.nxyz, N3S, d13.feat, 256), N2S, d32, 1);
  float* f2_3_2 = lin_f(c, interp_f(c, d22.nxyz, N2S, d23.nxyz, N3S, d23.feat, 256), N2S, d32, 1);
  float* c1_2 = concat2_f(c, d12.feat, 128, f1_3_2, 64, N2S);   // 192
  float* c2_2 = concat2_f(c, d22.feat, 128, f2_3_2, 64, N2S);
  float* f1_2_1 = lin_f(c, interp_f(c, d11.nxyz, N1S, d12.nxyz, N2S, d12.feat, 128), N1S, d21, 1);
  float* f2_2_1 = lin_f(c, interp_f(c, d21o.nxyz, N1S, d22.nxyz, N2S, d22.feat, 128), N1S, d21, 1);
  float* c1_1 = concat2_f(c, d11.feat, 64, f1_2_1, 32, N1S);    // 96
  float* c2_1 = concat2_f(c, d21o.feat, 64, f2_2_1, 32, N1S);
  float* f1_1_0 = lin_f(c, interp_f(c, xyz1, N, d11.nxyz, N1S, d11.feat, 64), N, d10, 1);
  float* f2_1_0 = lin_f(c, interp_f(c, xyz2, N, d21o.nxyz, N1S, d21o.feat, 64), N, d10, 1);
  float* c1_0 = concat2_f(c, f1_0, 32, f1_1_0, 32, N);          // 64
  float* c2_0 = concat2_f(c, f2_0, 32, f2_1_0, 32, N);

  // ---- level 2
  float* up_flow2 = interp_f(c, d12.nxyz, N2S, d13.nxyz, N3S, e3.f, 3);
  float* up_ff2   = interp_f(c, d12.nxyz, N2S, d13.nxyz, N3S, e3.ff, 3);
  float* pc2_2_w  = warp_f(c, d12.nxyz, N2S, d22.nxyz, N2S, up_flow2);
  float* cost2 = costvol_f(c, cv2, d12.nxyz, N2S, pc2_2_w, N2S, c1_2, c2_2, 192);
  float* feat3_up = interp_f(c, d12.nxyz, N2S, d13.nxyz, N3S, e3.feat, 64);
  float* nf1_2 = concat2_f(c, d12.feat, 128, feat3_up, 64, N2S);  // 192
  FEOut e2 = flow_est_f(c, fe2, d12.nxyz, N2S, nf1_2, 192, cost2, 128, up_flow2);
  float* flow2 = c.alloc((size_t)N2S * 3);
  add_kernel<<<cd((long)N2S * 3, 256), 256, 0, stream>>>(up_flow2, e2.f, flow2, (long)N2S * 3);
  float* ff2 = c.alloc((size_t)N2S * 3);
  add_kernel<<<cd((long)N2S * 3, 256), 256, 0, stream>>>(up_ff2, e2.ff, ff2, (long)N2S * 3);

  // ---- level 1
  float* up_flow1 = interp_f(c, d11.nxyz, N1S, d12.nxyz, N2S, flow2, 3);
  float* up_ff1   = interp_f(c, d11.nxyz, N1S, d12.nxyz, N2S, ff2, 3);
  float* pc2_1_w  = warp_f(c, d11.nxyz, N1S, d21o.nxyz, N1S, up_flow1);
  float* cost1 = costvol_f(c, cv1, d11.nxyz, N1S, pc2_1_w, N1S, c1_1, c2_1, 96);
  float* feat2_up = interp_f(c, d11.nxyz, N1S, d12.nxyz, N2S, e2.feat, 64);
  float* nf1_1 = concat2_f(c, d11.feat, 64, feat2_up, 64, N1S);   // 128
  FEOut e1 = flow_est_f(c, fe1, d11.nxyz, N1S, nf1_1, 128, cost1, 64, up_flow1);
  float* flow1 = c.alloc((size_t)N1S * 3);
  add_kernel<<<cd((long)N1S * 3, 256), 256, 0, stream>>>(up_flow1, e1.f, flow1, (long)N1S * 3);
  float* ff1 = c.alloc((size_t)N1S * 3);
  add_kernel<<<cd((long)N1S * 3, 256), 256, 0, stream>>>(up_ff1, e1.ff, ff1, (long)N1S * 3);

  // ---- level 0
  float* up_flow0 = interp_f(c, xyz1, N, d11.nxyz, N1S, flow1, 3);
  float* up_ff0   = interp_f(c, xyz1, N, d11.nxyz, N1S, ff1, 3);
  float* pc2_0_w  = warp_f(c, xyz1, N, xyz2, N, up_flow0);
  float* cost0 = costvol_f(c, cv0, xyz1, N, pc2_0_w, N, c1_0, c2_0, 64);
  float* feat1_up = interp_f(c, xyz1, N, d11.nxyz, N1S, e1.feat, 64);
  float* nf1_0 = concat2_f(c, f1_0, 32, feat1_up, 64, N);        // 96
  FEOut e0 = flow_est_f(c, fe0, xyz1, N, nf1_0, 96, cost0, 32, up_flow0);

  // ff0 = up_ff0 + rff0 -> output
  add_kernel<<<cd((long)N * 3, 256), 256, 0, stream>>>(up_ff0, e0.ff, (float*)d_out, (long)N * 3);
}
